// HyperGatedResidualAdapter_FFT_9414568313471
// MI455X (gfx1250) — compile-verified, matrix operand order checked by reading, not run on hardware
//
#include <hip/hip_runtime.h>
#include <hip/hip_bf16.h>
#include <math.h>

typedef __attribute__((ext_vector_type(16))) _Float16 v16h;
typedef __attribute__((ext_vector_type(8)))  float    v8f;

union F16x16 { v16h v; _Float16 a[16]; float4 q[2]; };
union F32x8  { v8f  v; float    a[8];  };
union H8pack { float4 q; _Float16 a[8]; };

#if defined(__HIP_DEVICE_COMPILE__) && \
    __has_builtin(__builtin_amdgcn_global_load_async_to_lds_b32) && \
    __has_builtin(__builtin_amdgcn_s_wait_asynccnt)
#define USE_ASYNC_LDS 1
#else
#define USE_ASYNC_LDS 0
#endif

// ---------------------------------------------------------------------------
// Workspace layout (floats):
//   [0   .. 192)  stats   [3][32][2]  (mean, std)
//   [192 .. 480)  bands   [3][32][3]  (power sums per FFT band)
//   [480 .. 512)  gamma   [32]
//   [512 .. 544)  beta    [32]
// ---------------------------------------------------------------------------

__global__ __launch_bounds__(256) void zero_ws_kernel(float* ws) {
    for (int i = threadIdx.x; i < 544; i += 256) ws[i] = 0.0f;
}

// ------------------------- per-image mean / std ----------------------------
__global__ __launch_bounds__(256) void stats_kernel(const float* __restrict__ nz,
                                                    const float* __restrict__ bo,
                                                    const float* __restrict__ mc,
                                                    float* __restrict__ ws) {
    int t = blockIdx.x / 32, b = blockIdx.x % 32;
    const float* src = (t == 0 ? nz : (t == 1 ? bo : mc)) + (size_t)b * 262144;
    float s = 0.f, ss = 0.f;
    for (int i = threadIdx.x; i < 262144; i += 256) {
        float v = src[i];
        s += v; ss += v * v;
    }
    __shared__ float rs[256], rss[256];
    rs[threadIdx.x] = s; rss[threadIdx.x] = ss;
    __syncthreads();
    for (int o = 128; o > 0; o >>= 1) {
        if (threadIdx.x < o) {
            rs[threadIdx.x]  += rs[threadIdx.x + o];
            rss[threadIdx.x] += rss[threadIdx.x + o];
        }
        __syncthreads();
    }
    if (threadIdx.x == 0) {
        const float N = 262144.f;
        float mean = rs[0] / N;
        float var  = (rss[0] - rs[0] * rs[0] / N) / (N - 1.f);   // ddof=1
        ws[(t * 32 + b) * 2 + 0] = mean;
        ws[(t * 32 + b) * 2 + 1] = sqrtf(fmaxf(var, 0.f));
    }
}

// --------------------- row rfft(512) power band sums -----------------------
__global__ __launch_bounds__(256) void fft_kernel(const float* __restrict__ nz,
                                                  const float* __restrict__ bo,
                                                  const float* __restrict__ mc,
                                                  float* __restrict__ ws) {
    __shared__ float2 Z[8][256];
    const int wave = threadIdx.x >> 5;
    const int lane = threadIdx.x & 31;
    const int R = blockIdx.x * 8 + wave;                 // 0 .. 49151
    const int t = R / 16384;
    const int rem = R % 16384;
    const int b = rem >> 9;
    const int h = rem & 511;
    const float* src = (t == 0 ? nz : (t == 1 ? bo : mc))
                       + (size_t)b * 262144 + (size_t)h * 512;

    // load with bit-reversed placement: z[n] = x[2n] + i x[2n+1]
    for (int j = 0; j < 8; ++j) {
        int n = lane + 32 * j;
        float re = src[2 * n], im = src[2 * n + 1];
        int rv = __brev((unsigned)n) >> 24;
        Z[wave][rv] = make_float2(re, im);
    }
    __syncthreads();

    // 8 radix-2 stages, 128 butterflies per stage (4 per lane)
    for (int s = 0; s < 8; ++s) {
        const int len = 1 << s;
        for (int j = 0; j < 4; ++j) {
            int k = lane + 32 * j;
            int pos = k & (len - 1);
            int i0 = ((k >> s) << (s + 1)) + pos;
            int i1 = i0 + len;
            float ang = -3.14159265358979f * (float)pos / (float)len;
            float sn, cs;
            __sincosf(ang, &sn, &cs);
            float2 a = Z[wave][i0], bb = Z[wave][i1];
            float2 tb = make_float2(bb.x * cs - bb.y * sn, bb.x * sn + bb.y * cs);
            Z[wave][i0] = make_float2(a.x + tb.x, a.y + tb.y);
            Z[wave][i1] = make_float2(a.x - tb.x, a.y - tb.y);
        }
        __syncthreads();
    }

    // real untangle: X[k] = Xe + e^{-i pi k/256} * Xo,  power into 3 bands
    float acc0 = 0.f, acc1 = 0.f, acc2 = 0.f;
    for (int j = 0; j < 8; ++j) {
        int k = lane + 32 * j;
        float2 zk = Z[wave][k];
        float2 zm = Z[wave][(256 - k) & 255];
        float xer = 0.5f * (zk.x + zm.x);
        float xei = 0.5f * (zk.y - zm.y);
        float xorr = 0.5f * (zk.y + zm.y);
        float xoi  = -0.5f * (zk.x - zm.x);
        float ang = -3.14159265358979f * (float)k * (1.f / 256.f);
        float sn, cs;
        __sincosf(ang, &sn, &cs);
        float Xr = xer + (xorr * cs - xoi * sn);
        float Xi = xei + (xorr * sn + xoi * cs);
        float p = Xr * Xr + Xi * Xi;
        if (k < 85) acc0 += p; else if (k < 170) acc1 += p; else acc2 += p;
        if (k == 0) {                       // Nyquist bin 256 (band 2)
            float xN = zk.x - zk.y;
            acc2 += xN * xN;
        }
    }
    for (int off = 16; off > 0; off >>= 1) {
        acc0 += __shfl_xor(acc0, off, 32);
        acc1 += __shfl_xor(acc1, off, 32);
        acc2 += __shfl_xor(acc2, off, 32);
    }
    if (lane == 0) {
        float* bnd = ws + 192 + (t * 32 + b) * 3;
        atomicAdd(&bnd[0], acc0);
        atomicAdd(&bnd[1], acc1);
        atomicAdd(&bnd[2], acc2);
    }
}

// --------------------------- hyper MLP (15->16->2) -------------------------
__global__ void mlp_kernel(const float* __restrict__ mw1, const float* __restrict__ mb1,
                           const float* __restrict__ mw2, const float* __restrict__ mb2,
                           float* __restrict__ ws) {
    int b = threadIdx.x;
    if (b >= 32) return;
    float feats[15];
    for (int t = 0; t < 3; ++t) {
        feats[2 * t + 0] = ws[(t * 32 + b) * 2 + 0];
        feats[2 * t + 1] = ws[(t * 32 + b) * 2 + 1];
    }
    const float bsz[3] = {85.f, 85.f, 87.f};
    for (int t = 0; t < 3; ++t) {
        float f[3]; float m3 = 0.f;
        for (int k = 0; k < 3; ++k) {
            float pm = ws[192 + (t * 32 + b) * 3 + k] / (512.f * bsz[k]);
            f[k] = log1pf(pm);
            m3 += f[k];
        }
        m3 *= (1.f / 3.f);
        for (int k = 0; k < 3; ++k)
            feats[6 + 3 * t + k] = f[k] / (m3 + 1e-6f);
    }
    float hid[16];
    for (int j = 0; j < 16; ++j) {
        float acc = mb1[j];
        for (int i = 0; i < 15; ++i) acc += mw1[j * 15 + i] * feats[i];
        hid[j] = fmaxf(acc, 0.f);
    }
    float hyp[2];
    for (int o = 0; o < 2; ++o) {
        float acc = mb2[o];
        for (int j = 0; j < 16; ++j) acc += mw2[o * 16 + j] * hid[j];
        hyp[o] = acc;
    }
    ws[480 + b] = 1.f / (1.f + expf(-hyp[0]));   // gamma
    ws[512 + b] = 0.1f * tanhf(hyp[1]);          // beta
}

// ---------------------------------------------------------------------------
// Fused conv1->conv2->conv3 + gated combine, WMMA implicit GEMM.
//
// 16-bit fragment K layout (wave32): lane m = L&15, half = L>>4;
// element e -> kk = kb*32 + 8*half + (e<8 ? e : 8+e).
// conv2/conv3 use tap-major K (kk = tap*16 + c) with channel-innermost f16
// hidden maps so each 8-element fragment half is one contiguous b128 LDS load
// at a compile-time constant offset from a single per-pixel base pointer
// (folds into ds_load_b128 immediate offsets).
// ---------------------------------------------------------------------------
__device__ __forceinline__ int klocal(int e, int half) {
    return e + (e >= 8 ? 8 : 0) + 8 * half;
}

__global__ __launch_bounds__(256) void conv_fused_kernel(
    const float* __restrict__ noisy, const float* __restrict__ base,
    const float* __restrict__ w1, const float* __restrict__ b1,
    const float* __restrict__ w2, const float* __restrict__ b2,
    const float* __restrict__ w3, const float* __restrict__ b3,
    const float* __restrict__ ws, float* __restrict__ out) {

    __shared__ float stageIn[2][22][22];                    // f32 async staging
    __shared__ _Float16 sIn[2][22][24];                     // f16 input halo
    __shared__ __align__(16) _Float16 sH1[20][22][16];      // [y][x][ch]
    __shared__ __align__(16) _Float16 sH2[18][20][16];      // [y][x][ch]

    const int img  = blockIdx.x >> 10;            // 1024 tiles per image
    const int tile = blockIdx.x & 1023;
    const int gy0 = (tile >> 5) * 16;
    const int gx0 = (tile & 31) * 16;

    const float* nz = noisy + (size_t)img * 262144;
    const float* bs = base  + (size_t)img * 262144;
    const float gamma = ws[480 + img];
    const float beta  = ws[512 + img];

    // ---- load input halo (22x22, 2 channels), zero-pad at borders ----
#if USE_ASYNC_LDS
    for (int idx = threadIdx.x; idx < 2 * 22 * 22; idx += 256) {
        int c = idx / 484, r = idx % 484;
        int iy = r / 22, ix = r % 22;
        int gy = gy0 - 3 + iy, gx = gx0 - 3 + ix;
        float* lp = &stageIn[c][iy][ix];
        if (gy >= 0 && gy < 512 && gx >= 0 && gx < 512) {
            const float* gp = (c ? bs : nz) + gy * 512 + gx;
            __builtin_amdgcn_global_load_async_to_lds_b32(
                (__attribute__((address_space(1))) int*)gp,
                (__attribute__((address_space(3))) int*)lp, 0, 0);
        } else {
            *lp = 0.f;
        }
    }
    __builtin_amdgcn_s_wait_asynccnt(0);
    __syncthreads();
    for (int idx = threadIdx.x; idx < 2 * 22 * 22; idx += 256) {
        int c = idx / 484, r = idx % 484;
        int iy = r / 22, ix = r % 22;
        sIn[c][iy][ix] = (_Float16)stageIn[c][iy][ix];
    }
#else
    for (int idx = threadIdx.x; idx < 2 * 22 * 22; idx += 256) {
        int c = idx / 484, r = idx % 484;
        int iy = r / 22, ix = r % 22;
        int gy = gy0 - 3 + iy, gx = gx0 - 3 + ix;
        float v = 0.f;
        if (gy >= 0 && gy < 512 && gx >= 0 && gx < 512)
            v = (c ? bs : nz)[gy * 512 + gx];
        sIn[c][iy][ix] = (_Float16)v;
        (void)stageIn;
    }
#endif
    __syncthreads();

    const int lane = threadIdx.x & 31;
    const int wave = threadIdx.x >> 5;
    const int m    = lane & 15;
    const int half = lane >> 4;
    const int hc   = 8 * half;

    // ================= conv1: 2ch -> 16ch, K=18 (c*9+tap order) =============
    F16x16 a1;
#pragma unroll
    for (int e = 0; e < 16; ++e) {
        int kk = klocal(e, half);
        a1.a[e] = (kk < 18) ? (_Float16)w1[m * 18 + kk] : (_Float16)0.f;
    }
    for (int nt = wave; nt < 25; nt += 8) {          // 400 px of 20x20
        int p = nt * 16 + m;
        int y1 = p / 20, x1 = p % 20;
        const _Float16* ip = &sIn[0][y1][x1];        // c stride 528, y stride 24
        F16x16 bf;
#pragma unroll
        for (int e = 0; e < 16; ++e) {
            int kk = klocal(e, half);
            if (kk < 18) {
                int c = kk / 9, tap = kk % 9;
                bf.a[e] = ip[c * 528 + (tap / 3) * 24 + (tap % 3)];
            } else bf.a[e] = (_Float16)0.f;
        }
        F32x8 d; d.v = (v8f){0.f,0.f,0.f,0.f,0.f,0.f,0.f,0.f};
        d.v = __builtin_amdgcn_wmma_f32_16x16x32_f16(false, a1.v, false, bf.v,
                                                     (short)0, d.v, false, false);
        H8pack st;
#pragma unroll
        for (int r = 0; r < 8; ++r)
            st.a[r] = (_Float16)fmaxf(d.a[r] + b1[r + hc], 0.f);
        *(float4*)&sH1[y1][x1][hc] = st.q;           // ds_store_b128
    }
    __syncthreads();

    // ============ conv2: 16ch -> 16ch, K=144 tap-major (tap*16+c) ==========
    F16x16 a2[5];
#pragma unroll
    for (int kb = 0; kb < 5; ++kb)
#pragma unroll
        for (int e = 0; e < 16; ++e) {
            int kk = kb * 32 + klocal(e, half);
            if (kk < 144) {
                int tap = kk >> 4, c = kk & 15;
                a2[kb].a[e] = (_Float16)w2[m * 144 + c * 9 + tap];
            } else a2[kb].a[e] = (_Float16)0.f;
        }
    for (int nt = wave; nt < 21; nt += 8) {          // 324 px of 18x18 (pad 336)
        int p = nt * 16 + m;
        bool valid = (p < 324);
        int pc = valid ? p : 0;
        int y2 = pc / 18, x2 = pc % 18;
        const _Float16* hp = &sH1[y2][x2][hc];       // tap off: (dy*22+dx)*16 elems
        F32x8 d; d.v = (v8f){0.f,0.f,0.f,0.f,0.f,0.f,0.f,0.f};
#pragma unroll
        for (int kb = 0; kb < 5; ++kb) {
            F16x16 bf;
            {   // fragment half 0: tap 2*kb
                const int t0 = 2 * kb;
                bf.q[0] = *(const float4*)(hp + ((t0 / 3) * 22 + (t0 % 3)) * 16);
            }
            if (kb < 4) {  // fragment half 1: tap 2*kb+1
                const int t1 = 2 * kb + 1;
                bf.q[1] = *(const float4*)(hp + ((t1 / 3) * 22 + (t1 % 3)) * 16);
            } else {
                bf.q[1] = make_float4(0.f, 0.f, 0.f, 0.f);
            }
            d.v = __builtin_amdgcn_wmma_f32_16x16x32_f16(false, a2[kb].v, false, bf.v,
                                                         (short)0, d.v, false, false);
        }
        if (valid) {
            H8pack st;
#pragma unroll
            for (int r = 0; r < 8; ++r)
                st.a[r] = (_Float16)fmaxf(d.a[r] + b2[r + hc], 0.f);
            *(float4*)&sH2[y2][x2][hc] = st.q;
        }
    }
    __syncthreads();

    // ====== conv3: 16ch -> 1ch (row M=0), K=144 tap-major + combine ========
    F16x16 a3[5];
#pragma unroll
    for (int kb = 0; kb < 5; ++kb)
#pragma unroll
        for (int e = 0; e < 16; ++e) {
            int kk = kb * 32 + klocal(e, half);
            if (m == 0 && kk < 144) {
                int tap = kk >> 4, c = kk & 15;
                a3[kb].a[e] = (_Float16)w3[c * 9 + tap];
            } else a3[kb].a[e] = (_Float16)0.f;
        }
    const float bias3 = b3[0];
    for (int nt = wave; nt < 16; nt += 8) {          // 256 px of 16x16
        int p = nt * 16 + m;
        int yo = p >> 4, xo = p & 15;
        const _Float16* hp = &sH2[yo][xo][hc];       // tap off: (dy*20+dx)*16 elems
        F32x8 d; d.v = (v8f){0.f,0.f,0.f,0.f,0.f,0.f,0.f,0.f};
#pragma unroll
        for (int kb = 0; kb < 5; ++kb) {
            F16x16 bf;
            {
                const int t0 = 2 * kb;
                bf.q[0] = *(const float4*)(hp + ((t0 / 3) * 20 + (t0 % 3)) * 16);
            }
            if (kb < 4) {
                const int t1 = 2 * kb + 1;
                bf.q[1] = *(const float4*)(hp + ((t1 / 3) * 20 + (t1 % 3)) * 16);
            } else {
                bf.q[1] = make_float4(0.f, 0.f, 0.f, 0.f);
            }
            d.v = __builtin_amdgcn_wmma_f32_16x16x32_f16(false, a3[kb].v, false, bf.v,
                                                         (short)0, d.v, false, false);
        }
        if (lane < 16) {                 // VGPR0 lanes 0-15 hold M=0 row
            int gy = gy0 + yo, gx = gx0 + xo;
            float r = d.a[0] + bias3;
            float bv = bs[gy * 512 + gx];
            float o = bv + gamma * r + beta;
            out[(size_t)img * 262144 + gy * 512 + gx] = fminf(fmaxf(o, 0.f), 1.f);
        }
    }
}

// ---------------------------------------------------------------------------
extern "C" void kernel_launch(void* const* d_in, const int* in_sizes, int n_in,
                              void* d_out, int out_size, void* d_ws, size_t ws_size,
                              hipStream_t stream) {
    const float* noisy = (const float*)d_in[0];
    const float* base  = (const float*)d_in[1];
    const float* mem   = (const float*)d_in[2];
    const float* w1 = (const float*)d_in[3];  const float* b1 = (const float*)d_in[4];
    const float* w2 = (const float*)d_in[5];  const float* b2 = (const float*)d_in[6];
    const float* w3 = (const float*)d_in[7];  const float* b3 = (const float*)d_in[8];
    const float* mw1 = (const float*)d_in[9];  const float* mb1 = (const float*)d_in[10];
    const float* mw2 = (const float*)d_in[11]; const float* mb2 = (const float*)d_in[12];
    float* out = (float*)d_out;
    float* ws  = (float*)d_ws;

    zero_ws_kernel<<<1, 256, 0, stream>>>(ws);
    stats_kernel<<<96, 256, 0, stream>>>(noisy, base, mem, ws);
    fft_kernel<<<6144, 256, 0, stream>>>(noisy, base, mem, ws);
    mlp_kernel<<<1, 32, 0, stream>>>(mw1, mb1, mw2, mb2, ws);
    conv_fused_kernel<<<32 * 1024, 256, 0, stream>>>(noisy, base,
                                                     w1, b1, w2, b2, w3, b3,
                                                     ws, out);
}
